// TriangleMultiplicationOutgoing_63531156242730
// MI455X (gfx1250) — matrix operand order repaired and checked by reading, not compiled, not run on hardware
//
#include <hip/hip_runtime.h>

#define N   512
#define NN  (N * N)
#define DP  128
#define EPS 1e-5f
#define LDP 520   // padded LDS row stride (elements): 1024B row + 16B TDM pad

typedef __attribute__((ext_vector_type(16))) __bf16       v16bf;
typedef __attribute__((ext_vector_type(8)))  float        v8f;
typedef __attribute__((ext_vector_type(4)))  unsigned int u32x4;
typedef __attribute__((ext_vector_type(2)))  unsigned int u32x2;
typedef __attribute__((ext_vector_type(8)))  int          i32x8;
typedef __attribute__((ext_vector_type(4)))  int          i32x4;

union FragU { v16bf bf; u32x4 u[2]; };
union Pack8 { u32x4 u; unsigned short s[8]; };
union Pack4 { u32x2 u; unsigned short s[4]; };

#if defined(__has_builtin)
#  if __has_builtin(__builtin_amdgcn_tensor_load_to_lds) && \
      __has_builtin(__builtin_amdgcn_s_wait_tensorcnt)
#    define USE_TDM 1
#  endif
#endif
#if defined(__has_include)
#  if __has_include(<hip/amd_detail/amd_gfx1250_TDM.h>)
#    define TDM_ARGS6 1
#  endif
#endif

__device__ __forceinline__ unsigned short f2bf(float f) {
  unsigned u = __float_as_uint(f);
  u += 0x7FFFu + ((u >> 16) & 1u);           // round-to-nearest-even
  return (unsigned short)(u >> 16);
}
__device__ __forceinline__ float bf2f(unsigned short s) {
  return __uint_as_float(((unsigned)s) << 16);
}
// v_exp_f32 + v_add + v_rcp_f32 (avoid the IEEE divide expansion)
__device__ __forceinline__ float sigmoidf(float x) {
  return __builtin_amdgcn_rcpf(1.0f + __expf(-x));
}

// A fragment (16x32 bf16) from row-major [M][ld] bf16 storage.
// Contiguous k order: lanes 0-15 M=0..15; VGPR0-3 hold K=kh..kh+7, VGPR4-7 K=kh+8..kh+15,
// kh = (lane>>4)*16  -> two contiguous 16-byte loads per lane.
__device__ __forceinline__ v16bf load_frag_a(const unsigned short* base, int ld, int lane) {
  int m  = lane & 15;
  int kh = (lane >> 4) << 4;
  FragU f;
  f.u[0] = *(const u32x4*)(base + m * ld + kh);
  f.u[1] = *(const u32x4*)(base + m * ld + kh + 8);
  return f.bf;
}

// B fragment (32x16 bf16), element (k,n) at base[n*ld + k] (B^T storage).
// Contiguous k order (as A): lanes 0-15 hold K=0..15, lanes 16-31 hold K=16..31.
__device__ __forceinline__ v16bf load_frag_bt(const unsigned short* base, int ld, int lane) {
  int n  = lane & 15;
  int k0 = (lane >> 4) << 4;
  FragU f;
  f.u[0] = *(const u32x4*)(base + n * ld + k0);
  f.u[1] = *(const u32x4*)(base + n * ld + k0 + 8);
  return f.bf;
}

// Pre-packed weight fragments: [tile][kstep][lane][16] bf16 -> one contiguous 32B read.
__device__ __forceinline__ v16bf load_wfrag(const unsigned short* wf, int tile, int s, int lane) {
  const unsigned short* p = wf + ((size_t)(((tile << 2) + s) << 5) + lane) * 16;
  FragU f;
  f.u[0] = *(const u32x4*)(p);
  f.u[1] = *(const u32x4*)(p + 8);
  return f.bf;
}

#ifdef USE_TDM
// Issue one TDM 2D tile load: tile_k x tile_rows bf16 elements, global row
// stride `ld` elements, into LDS offset lds_off. Optional LDS padding via the
// TDM pad engine (pad_bits ORed into group1 word0). D# per CDNA5 ISA 8.3-8.6.
__device__ __forceinline__ void tdm_load_tile(unsigned lds_off,
                                              const unsigned short* gptr,
                                              unsigned tile_k, unsigned tile_rows,
                                              unsigned ld, unsigned pad_bits) {
  unsigned long long ga = (unsigned long long)(size_t)gptr;
  u32x4 g0;
  g0[0] = 1u;                                   // count=1, user mode, no gather
  g0[1] = lds_off;                              // lds_addr (bytes)
  g0[2] = (unsigned)(ga & 0xFFFFFFFFu);         // global_addr[31:0]
  g0[3] = (unsigned)((ga >> 32) & 0x01FFFFFFu)  // global_addr[56:32]
          | (2u << 30);                         // type=2 ("image")
  i32x8 g1;
  g1[0] = (int)((1u << 16) | pad_bits);         // data_size=1 (2B) | pad config
  g1[1] = (int)((ld & 0xFFFFu) << 16);          // tensor_dim0[15:0]
  g1[2] = (int)((ld >> 16) | ((unsigned)N << 16));   // t_dim0 hi | t_dim1 lo
  g1[3] = (int)(tile_k << 16);                  // t_dim1 hi(0) | tile_dim0
  g1[4] = (int)tile_rows;                       // tile_dim1 | tile_dim2=0
  g1[5] = (int)ld;                              // tensor_dim0_stride[31:0]
  g1[6] = 0;                                    // stride hi | t_dim1_stride lo
  g1[7] = 0;
  i32x4 gz = {0, 0, 0, 0};
#ifdef TDM_ARGS6
  i32x8 gz8 = {0, 0, 0, 0, 0, 0, 0, 0};
  __builtin_amdgcn_tensor_load_to_lds(g0, g1, gz, gz, gz8, 0);
#else
  __builtin_amdgcn_tensor_load_to_lds(g0, g1, gz, gz, 0);
#endif
}
// pad_enable | pad_interval=7 (256 DWORDs = 1024B) | pad_amount=3 (4 DWORDs = 16B)
#define TDM_PAD_1024_16 ((1u << 20) | (7u << 22) | (3u << 25))
#endif

// ---------------------------------------------------------------------------
// Kernel 1: repack weights into WMMA B-fragment order (bf16).
// wfrag: 40 col-tiles x 4 ksteps x 32 lanes x 16 elems  (cols: L|Lg|R|Rg|Og)
// wofrag: 8 col-tiles x 4 ksteps x 32 x 16              (W_out)
// ---------------------------------------------------------------------------
__global__ __launch_bounds__(256) void k_prep(
    const float* __restrict__ Wl, const float* __restrict__ Wlg,
    const float* __restrict__ Wr, const float* __restrict__ Wrg,
    const float* __restrict__ Wog, const float* __restrict__ Wout,
    unsigned short* __restrict__ wfrag, unsigned short* __restrict__ wofrag) {
  int idx = blockIdx.x * 256 + threadIdx.x;
  const int NW = 40 * 4 * 32 * 16;
  if (idx < NW) {
    int e = idx & 15;
    int l = (idx >> 4) & 31;
    int s = (idx >> 9) & 3;
    int t = idx >> 11;
    int n = t * 16 + (l & 15);
    int k = s * 32 + ((l >> 4) << 4) + e;
    int sel = n >> 7, nc = n & 127;
    const float* W = (sel == 0) ? Wl : (sel == 1) ? Wlg : (sel == 2) ? Wr
                    : (sel == 3) ? Wrg : Wog;
    wfrag[idx] = f2bf(W[k * 128 + nc]);
  } else {
    int j = idx - NW;
    if (j < 8 * 4 * 32 * 16) {
      int e = j & 15;
      int l = (j >> 4) & 31;
      int s = (j >> 9) & 3;
      int t = j >> 11;
      int n = t * 16 + (l & 15);
      int k = s * 32 + ((l >> 4) << 4) + e;
      wofrag[j] = f2bf(Wout[k * 128 + n]);
    }
  }
}

// ---------------------------------------------------------------------------
// Kernel 2: fused LayerNorm + projections.
// Phase 1: wave-per-row LN of Z_raw (64 rows/block) -> bf16 LDS.
// Phase 2: [64x128]@[128x640] bf16 WMMA GEMM with gated epilogue.
// Writes left_t/right_t channel-major [c][i][k] bf16 and g_buf [row][d] bf16.
// ---------------------------------------------------------------------------
__global__ __launch_bounds__(256) void k_proj(
    const float* __restrict__ Zraw, const float* __restrict__ ln1_g,
    const float* __restrict__ ln1_b, const float* __restrict__ mask,
    const unsigned short* __restrict__ wfrag,
    const float* __restrict__ b_left, const float* __restrict__ b_lgate,
    const float* __restrict__ b_right, const float* __restrict__ b_rgate,
    const float* __restrict__ b_ogate,
    unsigned short* __restrict__ left_t, unsigned short* __restrict__ right_t,
    unsigned short* __restrict__ g_buf) {
  __shared__ unsigned short sZ[64 * 128];
  int tid = threadIdx.x;
  int wave = tid >> 5, lane = tid & 31;
  int row0 = blockIdx.x * 64;

  // ---- Phase 1: LayerNorm directly from Z_raw into the LDS A-tile ----
  const float4 g4 = *(const float4*)(ln1_g + lane * 4);
  const float4 b4 = *(const float4*)(ln1_b + lane * 4);
  for (int rr = 0; rr < 8; ++rr) {
    int rl = wave * 8 + rr;
    const float4 x = *(const float4*)(Zraw + (size_t)(row0 + rl) * DP + lane * 4);
    float s = x.x + x.y + x.z + x.w;
    for (int m = 16; m > 0; m >>= 1) s += __shfl_xor(s, m, 32);
    float mu = s * (1.0f / DP);
    float d0 = x.x - mu, d1 = x.y - mu, d2 = x.z - mu, d3 = x.w - mu;
    float q = d0 * d0 + d1 * d1 + d2 * d2 + d3 * d3;
    for (int m = 16; m > 0; m >>= 1) q += __shfl_xor(q, m, 32);
    float rstd = rsqrtf(q * (1.0f / DP) + EPS);
    Pack4 o;
    o.s[0] = f2bf(d0 * rstd * g4.x + b4.x);
    o.s[1] = f2bf(d1 * rstd * g4.y + b4.y);
    o.s[2] = f2bf(d2 * rstd * g4.z + b4.z);
    o.s[3] = f2bf(d3 * rstd * g4.w + b4.w);
    *(u32x2*)(sZ + rl * 128 + lane * 4) = o.u;
  }
  __syncthreads();

  // ---- Phase 2: WMMA GEMM + gated epilogue ----
  int n = lane & 15, hh = lane >> 4;
  for (int job = wave; job < 96; job += 8) {
    int rt = job / 24, jj = job % 24;
    const unsigned short* Ab = sZ + rt * 16 * 128;
    int rbase = row0 + rt * 16 + hh * 8;
    if (jj < 16) {
      bool isL = jj < 8;
      int qq = isL ? jj : (jj - 8);
      int t0 = isL ? qq : (16 + qq);   // raw tile
      int t1 = t0 + 8;                 // gate tile
      v8f acc0 = {0.f, 0.f, 0.f, 0.f, 0.f, 0.f, 0.f, 0.f};
      v8f acc1 = {0.f, 0.f, 0.f, 0.f, 0.f, 0.f, 0.f, 0.f};
      for (int s = 0; s < 4; ++s) {
        v16bf a = load_frag_a(Ab + 32 * s, 128, lane);
        v16bf w0 = load_wfrag(wfrag, t0, s, lane);
        v16bf w1 = load_wfrag(wfrag, t1, s, lane);
        acc0 = __builtin_amdgcn_wmma_f32_16x16x32_bf16(false, a, false, w0,
                                                       (short)0, acc0, false, false);
        acc1 = __builtin_amdgcn_wmma_f32_16x16x32_bf16(false, a, false, w1,
                                                       (short)0, acc1, false, false);
      }
      int c = qq * 16 + n;
      float braw = isL ? b_left[c] : b_right[c];
      float bgat = isL ? b_lgate[c] : b_rgate[c];
      Pack8 o;
      for (int v = 0; v < 8; ++v) {
        int row = rbase + v;
        float val = mask[row] * (acc0[v] + braw) * sigmoidf(acc1[v] + bgat);
        o.s[v] = f2bf(val);
      }
      unsigned short* dst = (isL ? left_t : right_t) + (size_t)c * NN + rbase;
      *(u32x4*)dst = o.u;
    } else {
      int qq = jj - 16;
      int t0 = 32 + qq;
      v8f acc = {0.f, 0.f, 0.f, 0.f, 0.f, 0.f, 0.f, 0.f};
      for (int s = 0; s < 4; ++s) {
        v16bf a = load_frag_a(Ab + 32 * s, 128, lane);
        v16bf w = load_wfrag(wfrag, t0, s, lane);
        acc = __builtin_amdgcn_wmma_f32_16x16x32_bf16(false, a, false, w,
                                                      (short)0, acc, false, false);
      }
      int d = qq * 16 + n;
      float bg = b_ogate[d];
      for (int v = 0; v < 8; ++v)
        g_buf[(size_t)(rbase + v) * 128 + d] = f2bf(sigmoidf(acc[v] + bg));
    }
  }
}

// ---------------------------------------------------------------------------
// Kernel 3: triangle einsum p[i,j,c] = sum_k left_t[c][i][k]*right_t[c][j][k].
// Batched over c: P_c = L_c @ R_c^T. 64x64 output tile per block. The FULL
// 64x512 L and R panels (130KB) are staged in dynamic LDS with a SINGLE
// Tensor Data Mover load each (pad engine inserts 16B per 1024B row ->
// 520-element padded stride, conflict-free fragment reads). Then 16 k-steps
// of pure ds_load + WMMA with no barriers in the hot loop.
// ---------------------------------------------------------------------------
__global__ __launch_bounds__(256) void k_tri(
    const unsigned short* __restrict__ left_t,
    const unsigned short* __restrict__ right_t, float* __restrict__ p_buf) {
  extern __shared__ __align__(16) unsigned short smem[];
  unsigned short* sL = smem;               // [64][LDP]
  unsigned short* sR = smem + 64 * LDP;    // [64][LDP]
  int i0 = blockIdx.x * 64, j0 = blockIdx.y * 64, c = blockIdx.z;
  const unsigned short* L = left_t + (size_t)c * NN + (size_t)i0 * N;
  const unsigned short* R = right_t + (size_t)c * NN + (size_t)j0 * N;
  int tid = threadIdx.x, wave = tid >> 5, lane = tid & 31;
  int st1 = wave + 8;
  int si0 = wave >> 2, sj0 = wave & 3;
  int si1 = st1 >> 2,  sj1 = st1 & 3;
#ifdef USE_TDM
  if (wave == 0) {
    tdm_load_tile((unsigned)(size_t)(void*)sL, L, N, 64, N, TDM_PAD_1024_16);
    tdm_load_tile((unsigned)(size_t)(void*)sR, R, N, 64, N, TDM_PAD_1024_16);
    __builtin_amdgcn_s_wait_tensorcnt(0);
  }
#else
  for (int e = tid; e < 64 * (N / 8); e += 256) {
    int rr = e >> 6, qq = (e & 63) * 8;
    *(u32x4*)(sL + rr * LDP + qq) = *(const u32x4*)(L + (size_t)rr * N + qq);
    *(u32x4*)(sR + rr * LDP + qq) = *(const u32x4*)(R + (size_t)rr * N + qq);
  }
#endif
  __syncthreads();
  v8f acc0 = {0.f, 0.f, 0.f, 0.f, 0.f, 0.f, 0.f, 0.f};
  v8f acc1 = {0.f, 0.f, 0.f, 0.f, 0.f, 0.f, 0.f, 0.f};
  const unsigned short* aB0 = sL + si0 * 16 * LDP;
  const unsigned short* aB1 = sL + si1 * 16 * LDP;
  const unsigned short* bB0 = sR + sj0 * 16 * LDP;
  const unsigned short* bB1 = sR + sj1 * 16 * LDP;
  for (int ks = 0; ks < 16; ++ks) {
    int k0 = ks * 32;
    v16bf a0 = load_frag_a(aB0 + k0, LDP, lane);
    v16bf a1 = load_frag_a(aB1 + k0, LDP, lane);
    v16bf b0 = load_frag_bt(bB0 + k0, LDP, lane);
    v16bf b1 = load_frag_bt(bB1 + k0, LDP, lane);
    acc0 = __builtin_amdgcn_wmma_f32_16x16x32_bf16(false, a0, false, b0,
                                                   (short)0, acc0, false, false);
    acc1 = __builtin_amdgcn_wmma_f32_16x16x32_bf16(false, a1, false, b1,
                                                   (short)0, acc1, false, false);
  }
  int n = lane & 15, hh = lane >> 4;
  for (int v = 0; v < 8; ++v) {
    int m = hh * 8 + v;
    int i = i0 + si0 * 16 + m, j = j0 + sj0 * 16 + n;
    p_buf[((size_t)i * N + j) * 128 + c] = acc0[v];
    i = i0 + si1 * 16 + m;
    j = j0 + sj1 * 16 + n;
    p_buf[((size_t)i * N + j) * 128 + c] = acc1[v];
  }
}

// ---------------------------------------------------------------------------
// Kernel 4: LN2(p) @ W_out + gated residual. 64 rows per block.
// ---------------------------------------------------------------------------
__global__ __launch_bounds__(256) void k_ln2_out(
    const float* __restrict__ p_buf, const float* __restrict__ g2,
    const float* __restrict__ b2, const unsigned short* __restrict__ wofrag,
    const float* __restrict__ out_bias, const unsigned short* __restrict__ g_buf,
    const float* __restrict__ Zraw, float* __restrict__ out) {
  __shared__ unsigned short sP[64 * 128];
  int tid = threadIdx.x, wave = tid >> 5, lane = tid & 31;
  int row0 = blockIdx.x * 64;
  const float4 gg = *(const float4*)(g2 + lane * 4);
  const float4 bb = *(const float4*)(b2 + lane * 4);
  for (int rr = 0; rr < 8; ++rr) {
    int rl = wave * 8 + rr;
    const float4 x = *(const float4*)(p_buf + (size_t)(row0 + rl) * 128 + lane * 4);
    float s = x.x + x.y + x.z + x.w;
    for (int m = 16; m > 0; m >>= 1) s += __shfl_xor(s, m, 32);
    float mu = s * (1.0f / 128.0f);
    float d0 = x.x - mu, d1 = x.y - mu, d2 = x.z - mu, d3 = x.w - mu;
    float q = d0 * d0 + d1 * d1 + d2 * d2 + d3 * d3;
    for (int m = 16; m > 0; m >>= 1) q += __shfl_xor(q, m, 32);
    float rstd = rsqrtf(q * (1.0f / 128.0f) + EPS);
    Pack4 o;
    o.s[0] = f2bf(d0 * rstd * gg.x + bb.x);
    o.s[1] = f2bf(d1 * rstd * gg.y + bb.y);
    o.s[2] = f2bf(d2 * rstd * gg.z + bb.z);
    o.s[3] = f2bf(d3 * rstd * gg.w + bb.w);
    *(u32x2*)(sP + rl * 128 + lane * 4) = o.u;
  }
  __syncthreads();
  int n = lane & 15, hh = lane >> 4;
  for (int t = 0; t < 4; ++t) {
    int tile = wave + 8 * t;
    int rt = tile >> 3, ct = tile & 7;
    v8f acc = {0.f, 0.f, 0.f, 0.f, 0.f, 0.f, 0.f, 0.f};
    for (int s = 0; s < 4; ++s) {
      v16bf a = load_frag_a(sP + rt * 16 * 128 + 32 * s, 128, lane);
      v16bf w = load_wfrag(wofrag, ct, s, lane);
      acc = __builtin_amdgcn_wmma_f32_16x16x32_bf16(false, a, false, w,
                                                    (short)0, acc, false, false);
    }
    int col = ct * 16 + n;
    float ob = out_bias[col];
    int rbase = row0 + rt * 16 + hh * 8;
    for (int v = 0; v < 8; ++v) {
      size_t idx = (size_t)(rbase + v) * 128 + col;
      float gv = bf2f(g_buf[idx]);
      out[idx] = Zraw[idx] + gv * (acc[v] + ob);
    }
  }
}

// ---------------------------------------------------------------------------
extern "C" void kernel_launch(void* const* d_in, const int* in_sizes, int n_in,
                              void* d_out, int out_size, void* d_ws, size_t ws_size,
                              hipStream_t stream) {
  (void)in_sizes; (void)n_in; (void)out_size; (void)ws_size;
  const float* Z_raw   = (const float*)d_in[0];
  const float* Z_mask  = (const float*)d_in[1];
  const float* ln1_g   = (const float*)d_in[2];
  const float* ln1_b   = (const float*)d_in[3];
  const float* W_left  = (const float*)d_in[4];
  const float* b_left  = (const float*)d_in[5];
  const float* W_right = (const float*)d_in[6];
  const float* b_right = (const float*)d_in[7];
  const float* W_lgate = (const float*)d_in[8];
  const float* b_lgate = (const float*)d_in[9];
  const float* W_rgate = (const float*)d_in[10];
  const float* b_rgate = (const float*)d_in[11];
  const float* W_ogate = (const float*)d_in[12];
  const float* b_ogate = (const float*)d_in[13];
  const float* ln2_g   = (const float*)d_in[14];
  const float* ln2_b   = (const float*)d_in[15];
  const float* W_out   = (const float*)d_in[16];
  const float* out_bias= (const float*)d_in[17];
  float* out = (float*)d_out;

  char* ws = (char*)d_ws;
  size_t off = 0;
  unsigned short* left_t  = (unsigned short*)(ws + off); off += (size_t)NN * 128 * 2;
  unsigned short* right_t = (unsigned short*)(ws + off); off += (size_t)NN * 128 * 2;
  unsigned short* g_buf   = (unsigned short*)(ws + off); off += (size_t)NN * 128 * 2;
  float*          p_buf   = (float*)(ws + off);          off += (size_t)NN * 128 * 4;
  unsigned short* wfrag   = (unsigned short*)(ws + off); off += (size_t)40 * 4 * 32 * 16 * 2;
  unsigned short* wofrag  = (unsigned short*)(ws + off); off += (size_t)8 * 4 * 32 * 16 * 2;

  k_prep<<<384, 256, 0, stream>>>(W_left, W_lgate, W_right, W_rgate, W_ogate,
                                  W_out, wfrag, wofrag);
  k_proj<<<NN / 64, 256, 0, stream>>>(Z_raw, ln1_g, ln1_b, Z_mask, wfrag,
                                      b_left, b_lgate, b_right, b_rgate,
                                      b_ogate, left_t, right_t, g_buf);
  dim3 gtri(N / 64, N / 64, 128);
  size_t tri_lds = (size_t)2 * 64 * LDP * 2;   // 133120 bytes dynamic LDS
  k_tri<<<gtri, 256, tri_lds, stream>>>(left_t, right_t, p_buf);
  k_ln2_out<<<NN / 64, 256, 0, stream>>>(p_buf, ln2_g, ln2_b, wofrag, out_bias,
                                         g_buf, Z_raw, out);
}
